// AKVMNGPCM_23751169147108
// MI455X (gfx1250) — matrix operand order repaired and checked by reading, not compiled, not run on hardware
//
#include <hip/hip_runtime.h>
#include <hip/hip_bf16.h>

// AKVMN-GPCM recurrent memory network for MI455X (gfx1250).
// Persistent workgroups: 16 batch rows each (batch independent; time sequential).
// Vm state (20.5MB) in d_ws => L2-resident (192MB L2). Weights staged to LDS/L2
// as bf16 once; dense GEMMs via v_wmma_f32_16x16x32_bf16 (f32 accumulate);
// recurrent state and epilogues stay f32. A-operand fragments register-cached
// across N-tiles; erase/add GEMMs fused to share A fragments; activations use
// v_exp_f32 + v_rcp_f32 fast paths (no IEEE division expansions).

#define NQ   1000
#define NC   4
#define MM   50
#define KK   50
#define VV   200
#define FF   50
#define BB   512
#define SS   500
#define TB   16     // batch rows per workgroup (WMMA M)
#define NTHR 256
#define NWAVES 8

typedef __attribute__((ext_vector_type(16))) __bf16 v16bf;
typedef __attribute__((ext_vector_type(8)))  float  v8f;

__device__ __host__ inline unsigned short f2bf_h(float f) {
  union { float f; unsigned u; } x; x.f = f;
  unsigned r = x.u + 0x7FFFu + ((x.u >> 16) & 1u);   // RNE
  return (unsigned short)(r >> 16);
}

struct Smem {
  float attn[TB][56];          // attention weights (f32, softmaxed)
  float feat[TB][56];          // final features
  float erase[TB][VV];         // sigmoid erase vector
  float addv[TB][VV];          // tanh add vector
  float biasq[64];             // zero-padded biases in LDS
  float bias1[112];
  float bias2[64];
  float biase[208];
  float biasa[208];
  int   qid[TB];
  int   rsp[TB];
  float maskv[TB];
  // bf16 operand arrays (zero-padded to WMMA multiples)
  unsigned short Wq[64 * 64];      // [n=out][k=in]
  unsigned short tk[64 * 64];      // transformed keys [m][k]
  unsigned short We[208 * 224];
  unsigned short Wa[208 * 224];
  unsigned short qeA[TB * 64];     // A: question embeds
  unsigned short qryA[TB * 64];    // A: tanh query
  unsigned short hA[TB * 256];     // A: [read(200) | qe(50) | pad]
  unsigned short hidA[TB * 128];   // A: relu hidden
  unsigned short veA[TB * 224];    // A: value embed
};

union AFrag { v16bf v; unsigned u[8]; };

__device__ inline float fast_rcp(float x) { return __builtin_amdgcn_rcpf(x); }

__device__ inline float act_apply(int ACT, float v) {
  if (ACT == 1) {                                       // tanh via exp+rcp
    float e = __expf(-2.f * v);
    v = (1.f - e) * fast_rcp(1.f + e);
  } else if (ACT == 2) {
    v = fmaxf(v, 0.f);                                  // relu
  } else if (ACT == 3) {
    v = fast_rcp(1.f + __expf(-v));                     // sigmoid
  }
  return v;
}

// Preload A(16xK) fragments into registers (shared across all N-tiles).
template <int KSTEPS>
__device__ inline void load_afrags(AFrag (&a)[KSTEPS], const unsigned short* A, int lda) {
  const int lane = threadIdx.x & 31;
  const int half = lane >> 4;
  const unsigned short* Arow = A + (lane & 15) * lda;
#pragma unroll
  for (int k = 0; k < KSTEPS; ++k) {
    const unsigned* Ap = (const unsigned*)(Arow + k * 32);
#pragma unroll
    for (int j = 0; j < 4; ++j) a[k].u[j]     = Ap[4 * half + j];        // K = 8*half+2j
#pragma unroll
    for (int j = 0; j < 4; ++j) a[k].u[4 + j] = Ap[8 + 4 * half + j];    // K = 16+8*half+2j
  }
}

template <int KSTEPS>
__device__ inline v8f wmma_ktile(const AFrag (&a)[KSTEPS],
                                 const unsigned short* Brow, int ldb) {
  const int lane = threadIdx.x & 31;
  const int half = lane >> 4;
  v8f acc = {0.f, 0.f, 0.f, 0.f, 0.f, 0.f, 0.f, 0.f};
#pragma unroll
  for (int k = 0; k < KSTEPS; ++k) {
    union { v16bf v; unsigned u[8]; } b;
    const unsigned* Bp = (const unsigned*)(Brow + k * 32);
#pragma unroll
    for (int j = 0; j < 8; ++j) b.u[j] = Bp[(j < 4 ? 0 : 4) + 4 * half + j];  // K = 8*half+2j (j<4), 16+8*half+2(j-4) (j>=4)
    acc = __builtin_amdgcn_wmma_f32_16x16x32_bf16(
        false, a[k].v, false, b.v, (short)0, acc, false, false);
  }
  return acc;
}

__device__ inline void epilogue(int ACT, v8f acc, int nt, int Nreal,
                                const float* bias,
                                float* outF, int ldoF,
                                unsigned short* outB, int ldoB) {
  const int lane = threadIdx.x & 31;
  const int half = lane >> 4;
  int n = nt * 16 + (lane & 15);
  float bconst = bias ? bias[n] : 0.f;    // LDS, zero-padded: unconditional load
  if (n < Nreal) {
#pragma unroll
    for (int r = 0; r < 8; ++r) {
      int m = r + 8 * half;
      float v = act_apply(ACT, acc[r] + bconst);
      if (outF) outF[m * ldoF + n] = v;
      if (outB) outB[m * ldoB + n] = f2bf_h(v);
    }
  }
}

// D = act(A * B^T + bias); B given as [N][K] rows. A fragments register-cached.
template <int ACT, int KSTEPS>
__device__ inline void gemm_tiles(const unsigned short* __restrict__ A, int lda,
                                  const unsigned short* __restrict__ B, int ldb,
                                  int Ntiles, int Nreal,
                                  const float* __restrict__ bias,
                                  float* __restrict__ outF, int ldoF,
                                  unsigned short* __restrict__ outB, int ldoB) {
  const int wave = threadIdx.x >> 5;
  AFrag a[KSTEPS];
  load_afrags<KSTEPS>(a, A, lda);
  for (int nt = wave; nt < Ntiles; nt += NWAVES) {
    const unsigned short* Brow = B + (nt * 16 + (threadIdx.x & 15)) * ldb;
    v8f acc = wmma_ktile<KSTEPS>(a, Brow, ldb);
    epilogue(ACT, acc, nt, Nreal, bias, outF, ldoF, outB, ldoB);
  }
}

// Fused erase/add: both GEMMs share A (ve), so share its register fragments.
// 26 wave-uniform tasks: task<13 -> sigmoid(We)+be -> erase; else tanh(Wa)+ba -> add.
template <int KSTEPS>
__device__ inline void gemm_erase_add(const unsigned short* __restrict__ A, int lda,
                                      const unsigned short* __restrict__ B0,
                                      const unsigned short* __restrict__ B1, int ldb,
                                      int Ntiles, int Nreal,
                                      const float* __restrict__ bias0,
                                      const float* __restrict__ bias1,
                                      float* __restrict__ out0,
                                      float* __restrict__ out1, int ldo) {
  const int wave = threadIdx.x >> 5;
  AFrag a[KSTEPS];
  load_afrags<KSTEPS>(a, A, lda);
  for (int task = wave; task < 2 * Ntiles; task += NWAVES) {
    const bool second = task >= Ntiles;          // wave-uniform
    const int nt = second ? task - Ntiles : task;
    const unsigned short* Brow =
        (second ? B1 : B0) + (nt * 16 + (threadIdx.x & 15)) * ldb;
    v8f acc = wmma_ktile<KSTEPS>(a, Brow, ldb);
    if (second) epilogue(1, acc, nt, Nreal, bias1, out1, ldo, nullptr, 0);
    else        epilogue(3, acc, nt, Nreal, bias0, out0, ldo, nullptr, 0);
  }
}

__global__ void stage_weights(const float* __restrict__ W1, const float* __restrict__ W2,
                              unsigned short* __restrict__ W1g,
                              unsigned short* __restrict__ W2g) {
  int tid = blockIdx.x * blockDim.x + threadIdx.x;
  int stride = gridDim.x * blockDim.x;
  for (int idx = tid; idx < 112 * 256; idx += stride) {       // [100][250] -> [112][256]
    int r = idx >> 8, c = idx & 255;
    W1g[idx] = f2bf_h((r < 100 && c < 250) ? W1[r * 250 + c] : 0.f);
  }
  for (int idx = tid; idx < 64 * 128; idx += stride) {        // [50][100] -> [64][128]
    int r = idx >> 7, c = idx & 127;
    W2g[idx] = f2bf_h((r < 50 && c < 100) ? W2[r * 100 + c] : 0.f);
  }
}

__global__ void __launch_bounds__(NTHR, 1)
akvmn_kernel(const int* __restrict__ questions, const int* __restrict__ responses,
             const float* __restrict__ q_embed, const float* __restrict__ Wv,
             const float* __restrict__ bv, const float* __restrict__ key_memory,
             const float* __restrict__ init_vm, const float* __restrict__ Wq,
             const float* __restrict__ bq, const float* __restrict__ Wk,
             const float* __restrict__ bk, const float* __restrict__ We,
             const float* __restrict__ be, const float* __restrict__ Wa,
             const float* __restrict__ ba, const float* __restrict__ b1,
             const float* __restrict__ b2, const float* __restrict__ Wo,
             const float* __restrict__ bo, const unsigned short* __restrict__ W1g,
             const unsigned short* __restrict__ W2g, float* __restrict__ Vm,
             float* __restrict__ out_feats, float* __restrict__ out_mastery,
             float* __restrict__ out_logits, float* __restrict__ out_probs) {
  extern __shared__ unsigned char smem_raw[];
  Smem& s = *reinterpret_cast<Smem*>(smem_raw);
  const int tid = threadIdx.x;
  const int b0  = blockIdx.x * TB;

  // ---- prologue: stage bf16 weights + biases to LDS, compute tk, init Vm ----
  for (int idx = tid; idx < 64 * 64; idx += NTHR) {
    int r = idx >> 6, c = idx & 63;
    s.Wq[idx] = f2bf_h((r < KK && c < KK) ? Wq[r * KK + c] : 0.f);
  }
  for (int idx = tid; idx < 64 * 64; idx += NTHR) {   // tk[m][k] = bk[m] + sum_i KM[i][k]*Wk[m][i]
    int m = idx >> 6, k = idx & 63;
    float v = 0.f;
    if (m < MM && k < KK) {
      v = bk[m];
      for (int i = 0; i < KK; ++i) v += key_memory[i * KK + k] * Wk[m * KK + i];
    }
    s.tk[idx] = f2bf_h(v);
  }
  for (int idx = tid; idx < 208 * 224; idx += NTHR) {
    int r = idx / 224, c = idx % 224;
    bool in = (r < VV && c < VV);
    s.We[idx] = f2bf_h(in ? We[r * VV + c] : 0.f);
    s.Wa[idx] = f2bf_h(in ? Wa[r * VV + c] : 0.f);
  }
  for (int idx = tid; idx < 112; idx += NTHR) s.bias1[idx] = (idx < 100) ? b1[idx] : 0.f;
  for (int idx = tid; idx < 64; idx += NTHR) {
    s.biasq[idx] = (idx < KK) ? bq[idx] : 0.f;
    s.bias2[idx] = (idx < FF) ? b2[idx] : 0.f;
  }
  for (int idx = tid; idx < 208; idx += NTHR) {
    s.biase[idx] = (idx < VV) ? be[idx] : 0.f;
    s.biasa[idx] = (idx < VV) ? ba[idx] : 0.f;
  }
  for (int idx = tid; idx < TB * MM * VV; idx += NTHR) {   // per-launch state init
    int i = idx / (MM * VV), rem = idx % (MM * VV);
    Vm[(size_t)(b0 + i) * (MM * VV) + rem] = init_vm[rem];
  }
  for (int idx = tid; idx < TB * 64; idx += NTHR)  { s.qeA[idx] = 0; s.qryA[idx] = 0; }
  for (int idx = tid; idx < TB * 256; idx += NTHR) s.hA[idx] = 0;
  for (int idx = tid; idx < TB * 128; idx += NTHR) s.hidA[idx] = 0;
  for (int idx = tid; idx < TB * 224; idx += NTHR) s.veA[idx] = 0;
  __syncthreads();

  // ---- sequential time loop; batch tile is private to this workgroup ----
  for (int t = 0; t < SS; ++t) {
    if (tid < TB) {
      int b = b0 + tid;
      int q = questions[b * SS + t];
      s.qid[tid] = q;
      s.rsp[tid] = responses[b * SS + t];
      s.maskv[tid] = (q > 0) ? 1.f : 0.f;
    }
    __syncthreads();
    // question embedding gather -> qeA and hA[:,200:250]
    for (int idx = tid; idx < TB * KK; idx += NTHR) {
      int i = idx / KK, k = idx % KK;
      unsigned short bfv = f2bf_h(q_embed[s.qid[i] * KK + k]);
      s.qeA[i * 64 + k] = bfv;
      s.hA[i * 256 + VV + k] = bfv;
    }
    __syncthreads();
    // query = tanh(qe @ Wq^T + bq)
    gemm_tiles<1, 2>(s.qeA, 64, s.Wq, 64, 4, KK, s.biasq, nullptr, 0, s.qryA, 64);
    __syncthreads();
    // scores = query @ tk^T
    gemm_tiles<0, 2>(s.qryA, 64, s.tk, 64, 4, MM, nullptr, &s.attn[0][0], 56, nullptr, 0);
    __syncthreads();
    // softmax over memory slots
    if (tid < TB) {
      float mx = -1e30f;
      for (int m = 0; m < MM; ++m) mx = fmaxf(mx, s.attn[tid][m]);
      float sum = 0.f;
      for (int m = 0; m < MM; ++m) { float e = __expf(s.attn[tid][m] - mx); s.attn[tid][m] = e; sum += e; }
      float inv = fast_rcp(sum);
      for (int m = 0; m < MM; ++m) s.attn[tid][m] *= inv;
    }
    __syncthreads();
    // read = attn @ Vm[b]  (per-row batched; Vm L2-resident) -> hA[:,0:200]
    for (int idx = tid; idx < TB * VV; idx += NTHR) {
      int i = idx / VV, v = idx % VV;
      const float* vm = Vm + (size_t)(b0 + i) * (MM * VV) + v;
      float acc = 0.f;
      for (int m = 0; m < MM; ++m) acc += s.attn[i][m] * vm[m * VV];
      s.hA[i * 256 + v] = f2bf_h(acc);
    }
    __syncthreads();
    // hidden = relu(h @ W1^T + b1); feat = hidden @ W2^T + b2
    gemm_tiles<2, 8>(s.hA, 256, W1g, 256, 7, 2 * FF, s.bias1, nullptr, 0, s.hidA, 128);
    __syncthreads();
    gemm_tiles<0, 4>(s.hidA, 128, W2g, 128, 4, FF, s.bias2, &s.feat[0][0], 56, nullptr, 0);
    __syncthreads();
    // outputs: feats, mastery, logits, probs
    for (int idx = tid; idx < TB * FF; idx += NTHR) {
      int i = idx / FF, f = idx % FF;
      out_feats[((size_t)(b0 + i) * SS + t) * FF + f] = s.feat[i][f];
    }
    if (tid < TB) {
      int b = b0 + tid;
      float msum = 0.f;
      for (int f = 0; f < FF; ++f) msum += s.feat[tid][f];
      out_mastery[(size_t)b * SS + t] = msum * (1.f / FF);
      float lg[NC]; float mx = -1e30f;
      for (int c = 0; c < NC; ++c) {
        float acc = bo[c];
        for (int f = 0; f < FF; ++f) acc += Wo[c * FF + f] * s.feat[tid][f];
        lg[c] = acc; mx = fmaxf(mx, acc);
      }
      float ex[NC]; float sum = 0.f;
      for (int c = 0; c < NC; ++c) { ex[c] = __expf(lg[c] - mx); sum += ex[c]; }
      float inv = fast_rcp(sum);
      size_t ob = ((size_t)b * SS + t) * NC;
      for (int c = 0; c < NC; ++c) { out_logits[ob + c] = lg[c]; out_probs[ob + c] = ex[c] * inv; }
    }
    // ve = mask * sum_c w_c * Wv[v, c*NQ + (q-1)] + bv   (GPCM linear-decay weights)
    for (int idx = tid; idx < TB * VV; idx += NTHR) {
      int i = idx / VV, v = idx % VV;
      int q = s.qid[i] - 1; q = q < 0 ? 0 : (q > NQ - 1 ? NQ - 1 : q);
      int r = s.rsp[i];
      const float* wv = Wv + (size_t)v * (NC * NQ) + q;
      float acc = 0.f;
#pragma unroll
      for (int c = 0; c < NC; ++c) {
        int d = c > r ? c - r : r - c;
        float w = (d == 0) ? 1.f : (d == 1) ? (2.f / 3.f) : (d == 2) ? (1.f / 3.f) : 0.f;
        acc += w * wv[c * NQ];
      }
      s.veA[i * 224 + v] = f2bf_h(acc * s.maskv[i] + bv[v]);
    }
    __syncthreads();
    // erase = sigmoid(ve @ We^T + be); add = tanh(ve @ Wa^T + ba)  [fused, shared A]
    gemm_erase_add<7>(s.veA, 224, s.We, s.Wa, 224, 13, VV,
                      s.biase, s.biasa, &s.erase[0][0], &s.addv[0][0], VV);
    __syncthreads();
    // Vm = Vm*(1 - attn*erase) + attn*add   (L2-resident RMW)
    for (int idx = tid; idx < TB * MM * VV; idx += NTHR) {
      int i = idx / (MM * VV), rem = idx % (MM * VV);
      int m = rem / VV, v = rem % VV;
      float at = s.attn[i][m];
      size_t g = (size_t)(b0 + i) * (MM * VV) + rem;
      float old = Vm[g];
      Vm[g] = old * (1.f - at * s.erase[i][v]) + at * s.addv[i][v];
    }
    __syncthreads();
  }
}

extern "C" void kernel_launch(void* const* d_in, const int* in_sizes, int n_in,
                              void* d_out, int out_size, void* d_ws, size_t ws_size,
                              hipStream_t stream) {
  (void)in_sizes; (void)n_in; (void)out_size; (void)ws_size;
  const int*   questions  = (const int*)d_in[0];
  const int*   responses  = (const int*)d_in[1];
  const float* q_embed    = (const float*)d_in[2];
  const float* Wv         = (const float*)d_in[3];
  const float* bv         = (const float*)d_in[4];
  const float* key_memory = (const float*)d_in[5];
  const float* init_vm    = (const float*)d_in[6];
  const float* Wq         = (const float*)d_in[7];
  const float* bq         = (const float*)d_in[8];
  const float* Wk         = (const float*)d_in[9];
  const float* bk         = (const float*)d_in[10];
  const float* We         = (const float*)d_in[11];
  const float* be         = (const float*)d_in[12];
  const float* Wa         = (const float*)d_in[13];
  const float* ba         = (const float*)d_in[14];
  const float* W1         = (const float*)d_in[15];
  const float* b1         = (const float*)d_in[16];
  const float* W2         = (const float*)d_in[17];
  const float* b2         = (const float*)d_in[18];
  const float* Wo         = (const float*)d_in[19];
  const float* bo         = (const float*)d_in[20];

  // workspace: Vm f32 state (20.48 MB, L2-resident), then bf16 W1/W2 copies
  float* Vm = (float*)d_ws;
  unsigned short* W1g = (unsigned short*)((char*)d_ws + (size_t)BB * MM * VV * 4);
  unsigned short* W2g = W1g + 112 * 256;

  float* out         = (float*)d_out;
  float* out_feats   = out;                                   // [B,S,F]
  float* out_mastery = out + (size_t)BB * SS * FF;            // [B,S]
  float* out_logits  = out_mastery + (size_t)BB * SS;         // [B,S,C]
  float* out_probs   = out_logits + (size_t)BB * SS * NC;     // [B,S,C]

  stage_weights<<<64, NTHR, 0, stream>>>(W1, W2, W1g, W2g);
  akvmn_kernel<<<BB / TB, NTHR, sizeof(Smem), stream>>>(
      questions, responses, q_embed, Wv, bv, key_memory, init_vm,
      Wq, bq, Wk, bk, We, be, Wa, ba, b1, b2, Wo, bo,
      W1g, W2g, Vm, out_feats, out_mastery, out_logits, out_probs);
}
